// MultiStageAggregateTransformer_53188874994221
// MI455X (gfx1250) — matrix operand order repaired and checked by reading, not compiled, not run on hardware
//
#include <hip/hip_runtime.h>
#include <hip/hip_bf16.h>
#include <math.h>

typedef __attribute__((ext_vector_type(16))) _Float16 v16h;
typedef __attribute__((ext_vector_type(8)))  float    v8f;

#define NUM_CLIP 256
#define NBOX 8192
#define NBATCH 32
#define WORDS_PER_ROW (NBOX / 32)   // 256
#define COLW_PER_BLOCK 4            // 128 columns (8 WMMAs) per wave

// ---------------- Kernel 1: gt_dist (B x 256 x 3 Gaussians) ----------------
__global__ void gt_dist_kernel(const float* __restrict__ gt, float* __restrict__ out) {
  int t = blockIdx.x * blockDim.x + threadIdx.x;
  if (t >= NBATCH * NUM_CLIP * 3) return;
  int j = t % 3;
  int c = (t / 3) % NUM_CLIP;
  int b = t / (3 * NUM_CLIP);
  float g0 = gt[2 * b], g1 = gt[2 * b + 1];
  float mid = 0.5f * (g0 + g1);
  float x = (j == 0 ? g0 : (j == 1 ? g1 : mid)) * (float)NUM_CLIP;
  float alpha = (j == 2) ? 0.21f : 0.25f;
  float len = (g1 - g0) * (float)NUM_CLIP;
  float sigma = alpha * len;
  float d = (float)c - x;
  out[t] = expf(-d * d / (2.0f * sigma * sigma));
}

// ---------------- Kernel 2: LDS-resident bitonic argsort (descending score) -
// 8192 keys + 8192 payload indices = 64KB LDS, one workgroup (32 waves).
// Tail also materializes the f16 one-hot batch matrix (8192 x 32, 512KB)
// in sorted order so the WMMA kernel can load A/B fragments directly.
__global__ __launch_bounds__(1024) void sort_kernel(
    const float* __restrict__ pred, const float* __restrict__ score,
    const int* __restrict__ bidx,
    float* __restrict__ s_out, float* __restrict__ e_out,
    float* __restrict__ sc_out, unsigned* __restrict__ oh /* 16 words/box */) {
  __shared__ float sk[NBOX];
  __shared__ int   si[NBOX];
  int tid = threadIdx.x;
  for (int i = tid; i < NBOX; i += 1024) { sk[i] = score[i]; si[i] = i; }
  __syncthreads();
  for (int k = 2; k <= NBOX; k <<= 1) {
    for (int j = k >> 1; j > 0; j >>= 1) {
      for (int i = tid; i < NBOX; i += 1024) {
        int ixj = i ^ j;
        if (ixj > i) {
          float a = sk[i], bval = sk[ixj];
          int ia = si[i], ib = si[ixj];
          // higher score first; stable tie-break on original index
          bool before = (bval > a) || (bval == a && ib < ia);
          bool up = ((i & k) == 0);
          if (before == up) { sk[i] = bval; sk[ixj] = a; si[i] = ib; si[ixj] = ia; }
        }
      }
      __syncthreads();
    }
  }
  for (int i = tid; i < NBOX; i += 1024) {
    int o = si[i];
    s_out[i]  = pred[2 * o];
    e_out[i]  = pred[2 * o + 1];
    sc_out[i] = sk[i];
    // one-hot row: 32 f16 (16 dwords); f16 1.0 = 0x3C00
    unsigned bm = 1u << bidx[o];
    unsigned w[16];
#pragma unroll
    for (int p = 0; p < 16; ++p)
      w[p] = (((bm >> (2 * p))     & 1u) ? 0x00003C00u : 0u) |
             (((bm >> (2 * p + 1)) & 1u) ? 0x3C000000u : 0u);
    uint4* dst = (uint4*)(oh + (size_t)i * 16);
    dst[0] = make_uint4(w[0],  w[1],  w[2],  w[3]);
    dst[1] = make_uint4(w[4],  w[5],  w[6],  w[7]);
    dst[2] = make_uint4(w[8],  w[9],  w[10], w[11]);
    dst[3] = make_uint4(w[12], w[13], w[14], w[15]);
  }
}

// ---------------- Kernel 3: suppression bit-matrix via WMMA one-hot GEMM ----
// One wave per 16-row x 128-col tile. Batch equality = onehot(b_i).onehot(b_j)
// on the matrix pipe (V_WMMA_F32_16X16X32_F16, K=32=NBATCH, exact 0/1);
// fragments are straight b128 loads from the precomputed L2-resident table;
// division-free IoU threshold on the VALU co-executes; bits gathered with
// wave32 ballots (no LDS, no EXEC save/restore in the inner loop).
__global__ __launch_bounds__(32) void mask_kernel(
    const float* __restrict__ s, const float* __restrict__ e,
    const _Float16* __restrict__ oh, unsigned* __restrict__ mask) {
  int w0   = blockIdx.x * COLW_PER_BLOCK;   // first column word (32 cols each)
  int i0   = blockIdx.y * 16;               // first row
  int lane = threadIdx.x;
  int hi = lane >> 4, lo = lane & 15;

  // A fragment 16x32 f16: lane row M=lo; elems t hold
  // K = hi*16 + t (contiguous, matching B).
  union AF { float4 f4[2]; v16h h; } af;
  const _Float16* arow = oh + (size_t)(i0 + lo) * 32;
  af.f4[0] = *(const float4*)(arow + hi * 16);
  af.f4[1] = *(const float4*)(arow + hi * 16 + 8);
  v16h a = af.h;

  // Row intervals for this lane's 8 output rows (m = hi*8 + r), as b128 loads.
  const float4* s4 = (const float4*)(s + i0 + hi * 8);
  const float4* e4 = (const float4*)(e + i0 + hi * 8);
  float4 sA = s4[0], sB = s4[1], eA = e4[0], eB = e4[1];
  float rs[8]  = {sA.x, sA.y, sA.z, sA.w, sB.x, sB.y, sB.z, sB.w};
  float re_[8] = {eA.x, eA.y, eA.z, eA.w, eB.x, eB.y, eB.z, eB.w};

  unsigned myw[COLW_PER_BLOCK] = {0u, 0u, 0u, 0u};  // lane<16 owns row = lane

#pragma unroll
  for (int hf = 0; hf < 2 * COLW_PER_BLOCK; ++hf) {
    int w     = hf >> 1;
    int shift = (hf & 1) * 16;
    int jc    = (w0 + w) * 32 + (hf & 1) * 16;

    // B fragment 32x16 f16: lane = column N, elems t -> K = t + 16*hi:
    // one contiguous 32B chunk of the one-hot row.
    v16h bb = *(const v16h*)(oh + (size_t)(jc + lo) * 32 + hi * 16);

    v8f acc = {};
    acc = __builtin_amdgcn_wmma_f32_16x16x32_f16(
        false, a, false, bb, (short)0, acc, false, false);

    float sj = s[jc + lo], ej = e[jc + lo];
    float lj = ej - sj;

#pragma unroll
    for (int r = 0; r < 8; ++r) {
      float inter = fmaxf(fminf(re_[r], ej) - fmaxf(rs[r], sj), 0.0f);
      float uni = (re_[r] - rs[r]) + lj - inter;
      // iou > 0.5  <=>  inter > 0.5*max(union,1e-8)   (denominator > 0)
      bool pred = (acc[r] > 0.5f) && (inter > 0.5f * fmaxf(uni, 1e-8f));
      unsigned bal = __builtin_amdgcn_ballot_w32(pred);
      // bits 0..15: (row r, col n); bits 16..31: (row r+8, col n)
      myw[w] |= (lane == r)     ? ((bal & 0xFFFFu) << shift) : 0u;
      myw[w] |= (lane == r + 8) ? ((bal >> 16)     << shift) : 0u;
    }
  }

  if (lane < 16) {
#pragma unroll
    for (int w = 0; w < COLW_PER_BLOCK; ++w)
      mask[(size_t)(i0 + lane) * WORDS_PER_ROW + (w0 + w)] = myw[w];
  }
}

// ---------------- Kernel 4: sequential bitmask NMS scan + masked outputs ----
__global__ __launch_bounds__(256) void scan_kernel(
    const unsigned* __restrict__ mask,
    const float* __restrict__ s, const float* __restrict__ e,
    const float* __restrict__ sc,
    float* __restrict__ out_bds, float* __restrict__ out_sc) {
  __shared__ unsigned keep[WORDS_PER_ROW];
  int t = threadIdx.x;
  keep[t] = 0xFFFFFFFFu;
  __syncthreads();
  for (int i = 0; i < NBOX; ++i) {
    unsigned alive = (keep[i >> 5] >> (i & 31)) & 1u;
    __syncthreads();                 // all reads before any write this round
    if (alive) {
      unsigned w = mask[(size_t)i * WORDS_PER_ROW + t];
      int iw = i >> 5;
      if (t < iw)       w = 0u;                                  // j < i
      else if (t == iw) w &= ~(0xFFFFFFFFu >> (31 - (i & 31)));  // j <= i
      keep[t] &= ~w;
    }
    __syncthreads();
  }
  unsigned kw = keep[t];
  for (int bpos = 0; bpos < 32; ++bpos) {
    int k = t * 32 + bpos;
    float f = ((kw >> bpos) & 1u) ? 1.0f : 0.0f;
    out_bds[2 * k]     = s[k] * f;
    out_bds[2 * k + 1] = e[k] * f;
    out_sc[k]          = sc[k] * f;
  }
}

extern "C" void kernel_launch(void* const* d_in, const int* in_sizes, int n_in,
                              void* d_out, int out_size, void* d_ws, size_t ws_size,
                              hipStream_t stream) {
  const float* gt    = (const float*)d_in[0];   // (32, 2)
  const float* pred  = (const float*)d_in[1];   // (8192, 2)
  const float* score = (const float*)d_in[2];   // (8192,)
  const int*   bidx  = (const int*)d_in[3];     // (8192,)

  float* out      = (float*)d_out;
  float* out_gt   = out;                            // 32*256*3 = 24576
  float* out_bds  = out + NBATCH * NUM_CLIP * 3;    // 8192*2   = 16384
  float* out_sc   = out_bds + 2 * NBOX;             // 8192

  char* ws = (char*)d_ws;
  float*    s_s  = (float*)(ws);                        // 32 KB
  float*    e_s  = (float*)(ws + 4 * NBOX);             // 32 KB
  float*    sc_s = (float*)(ws + 8 * NBOX);             // 32 KB
  unsigned* oh   = (unsigned*)(ws + 12 * NBOX);         // 8192*64B = 512 KB
  unsigned* mask = (unsigned*)(ws + 12 * NBOX + 64 * NBOX); // 8 MB

  gt_dist_kernel<<<(NBATCH * NUM_CLIP * 3 + 255) / 256, 256, 0, stream>>>(gt, out_gt);
  sort_kernel<<<1, 1024, 0, stream>>>(pred, score, bidx, s_s, e_s, sc_s, oh);
  dim3 grid(WORDS_PER_ROW / COLW_PER_BLOCK, NBOX / 16);
  mask_kernel<<<grid, 32, 0, stream>>>(s_s, e_s, (const _Float16*)oh, mask);
  scan_kernel<<<1, 256, 0, stream>>>(mask, s_s, e_s, sc_s, out_bds, out_sc);
}
